// MorphologyNeuralNet_83193516523625
// MI455X (gfx1250) — matrix operand order repaired and checked by reading, not compiled, not run on hardware
//
#include <hip/hip_runtime.h>
#include <math.h>

// ---------------------------------------------------------------------------
// MorphologyNeuralNet for gfx1250 (MI455X, wave32, WMMA)
//
//   1) k_init        : slots init
//   2) k_meanpool    : 512 blocks -> deterministic partial token sums
//   3) k_reduce_feat : feat = (sum of 8 partials) / 1024
//   4) k_mlp1        : 32 single-wave blocks, v_wmma_f32_16x16x32_f16,
//                      GELU -> f16 hidden state in ws
//   5) k_mlp2        : 1 block / 4 waves, WMMA, -> x[16][64]
//   6) k_params      : SE weights e^{w}, bias, sign
//   7) k_minmax      : per-batch input range (ordered-uint atomics)
//   8) k_prep        : E = exp(s*H - Xmax)
//   9) 3x { k_conv   : Z = s*(log(corr7x7(E,e^w)) + Xmax + bias) + range atomics
//           k_norm   : renormalize -> next E / final output }
// ---------------------------------------------------------------------------

typedef __attribute__((ext_vector_type(16))) _Float16 v16h;
typedef __attribute__((ext_vector_type(8)))  float    v8f;

#define BATCH 16
#define NP    1024
#define DIM   256
#define CDIM  768
#define INCH  1024
#define HID   512
#define OUTP  51
#define HW    256
#define NPIX  (HW*HW)        // 65536
#define KS    49
#define NSPLIT 8             // token splits for mean-pool

// monotone float <-> uint encoding for unsigned-int min/max atomics
__device__ __forceinline__ unsigned f2o(float f) {
  unsigned u = __float_as_uint(f);
  return (u >> 31) ? ~u : (u ^ 0x80000000u);
}
__device__ __forceinline__ float o2f(unsigned u) {
  return __uint_as_float((u >> 31) ? (u ^ 0x80000000u) : ~u);
}

// ---------------------------------------------------------------------------
__global__ void k_init(unsigned* slots) {
  int t = threadIdx.x;
  if (t < 128) slots[t] = (t & 1) ? 0u : 0xFFFFFFFFu;
}

// ---------------------------------------------------------------------------
// mean pool partials: grid (16, 4, NSPLIT), block 256.  Each block sums
// NP/NSPLIT tokens for 256 consecutive channels -> fully coalesced; 512
// blocks keep the 64MB stream at HBM rate.  Fixed-order sums => deterministic.
__global__ __launch_bounds__(256) void k_meanpool(const float* __restrict__ mf,
                                                  const float* __restrict__ pf,
                                                  float* __restrict__ partial) {
  int b = blockIdx.x, g = blockIdx.y, sp = blockIdx.z, tid = threadIdx.x;
  int n0 = sp * (NP / NSPLIT);
  float s = 0.0f;
  int c;
  if (g == 0) {
    c = tid;
    const float* p = mf + (size_t)b * NP * DIM + (size_t)n0 * DIM + tid;
    #pragma unroll 4
    for (int n = 0; n < NP / NSPLIT; ++n) s += p[(size_t)n * DIM];
  } else {
    int cc = (g - 1) * 256 + tid;               // 0..767
    c = 256 + cc;
    const float* p = pf + (size_t)b * NP * CDIM + (size_t)n0 * CDIM + cc;
    #pragma unroll 4
    for (int n = 0; n < NP / NSPLIT; ++n) s += p[(size_t)n * CDIM];
  }
  partial[(size_t)sp * BATCH * INCH + b * INCH + c] = s;
}

// feat = (sum over splits) / NP.  grid 64, block 256.
__global__ __launch_bounds__(256) void k_reduce_feat(const float* __restrict__ partial,
                                                     float* __restrict__ feat) {
  int i = blockIdx.x * 256 + threadIdx.x;       // 0..16383
  float s = 0.0f;
  #pragma unroll
  for (int sp = 0; sp < NSPLIT; ++sp) s += partial[(size_t)sp * BATCH * INCH + i];
  feat[i] = s * (1.0f / (float)NP);
}

// ---------------------------------------------------------------------------
// MLP layer 1: 32 blocks x 32 threads (1 wave32 each, EXEC all ones).
// Block nt computes the 16x16 tile [0:16, nt*16:(nt+1)*16] of
// GELU(feat @ W1 + b1) and stores it as f16 into hbuf[16][512].
// WMMA operands use contiguous k = 16*(lane/16) + e in both A and B.
__global__ __launch_bounds__(32) void k_mlp1(const float* __restrict__ feat,
                                             const float* __restrict__ W1,
                                             const float* __restrict__ b1,
                                             _Float16* __restrict__ hbuf) {
  __shared__ _Float16 sA[16 * INCH];   // 32 KB
  int lane = threadIdx.x;
  int m  = lane & 15;
  int lh = lane >> 4;
  int n0 = blockIdx.x * 16;

  for (int i = lane; i < 16 * INCH; i += 32) sA[i] = (_Float16)feat[i];
  __syncthreads();

  v8f acc = {};
  for (int kk = 0; kk < INCH / 32; ++kk) {
    int kb = kk * 32;
    v16h a, bm;
    #pragma unroll
    for (int e = 0; e < 16; ++e) {
      int ka = kb + e + 16 * lh;   // A: 16x32 f16 layout
      a[e] = sA[m * INCH + ka];
      int kbm = kb + e + 16 * lh;                           // B: lanes0-15 K=0..15
      bm[e] = (_Float16)W1[(size_t)kbm * HID + n0 + m];
    }
    acc = __builtin_amdgcn_wmma_f32_16x16x32_f16(
        false, a, false, bm, (short)0, acc, false, false);
  }
  int n = n0 + m;
  float bv = b1[n];
  #pragma unroll
  for (int r = 0; r < 8; ++r) {
    int row = r + 8 * lh;                                   // C/D row layout
    float v = acc[r] + bv;
    v = 0.5f * v * (1.0f + erff(v * 0.70710678118654752f)); // exact GELU
    hbuf[row * HID + n] = (_Float16)v;
  }
}

// MLP layer 2: 1 block, 128 threads = 4 waves; N=51 padded to 64 -> 4 tiles.
__global__ __launch_bounds__(128) void k_mlp2(const _Float16* __restrict__ hbuf,
                                              const float* __restrict__ W2,
                                              const float* __restrict__ b2,
                                              float* __restrict__ x) {
  __shared__ _Float16 sH[16 * HID];    // 16 KB
  int tid  = threadIdx.x;
  int lane = tid & 31;
  int wave = tid >> 5;
  int m  = lane & 15;
  int lh = lane >> 4;

  for (int i = tid; i < 16 * HID; i += 128) sH[i] = hbuf[i];
  __syncthreads();

  int n0 = wave * 16;
  int n  = n0 + m;
  v8f acc = {};
  for (int kk = 0; kk < HID / 32; ++kk) {
    int kb = kk * 32;
    v16h a, bm;
    #pragma unroll
    for (int e = 0; e < 16; ++e) {
      int ka = kb + e + 16 * lh;
      a[e] = sH[m * HID + ka];
      int kbm = kb + e + 16 * lh;
      bm[e] = (n < OUTP) ? (_Float16)W2[(size_t)kbm * OUTP + n] : (_Float16)0.0f;
    }
    acc = __builtin_amdgcn_wmma_f32_16x16x32_f16(
        false, a, false, bm, (short)0, acc, false, false);
  }
  if (n < OUTP) {
    float bv = b2[n];
    #pragma unroll
    for (int r = 0; r < 8; ++r) {
      int row = r + 8 * lh;
      x[row * 64 + n] = acc[r] + bv;
    }
  }
}

// ---------------------------------------------------------------------------
// per-batch params: pr[b][0..48] = exp(0.5*(sig(x_k)+sig(x_{48-k}))),
// pr[b][49] = bias, pr[b][50] = s_eff (tanh, or 1 if |s|<eps)
__global__ void k_params(const float* __restrict__ x, float* __restrict__ pr) {
  int b = blockIdx.x, t = threadIdx.x;
  __shared__ float sig[KS];
  const float* xb = x + b * 64;
  if (t < KS) sig[t] = 1.0f / (1.0f + expf(-xb[t]));
  __syncthreads();
  float* prb = pr + b * 52;
  if (t < KS)  prb[t]  = expf(0.5f * (sig[t] + sig[KS - 1 - t]));  // e^{w_k}
  if (t == 49) prb[49] = xb[49];                                   // bias
  if (t == 50) {
    float s = tanhf(xb[50]);
    prb[50] = (fabsf(s) >= 1e-7f) ? s : 1.0f;                      // s_eff
  }
}

// ---------------------------------------------------------------------------
// per-batch min/max of H into slots (stage base). grid (16,16), block 256.
__global__ __launch_bounds__(256) void k_minmax(const float* __restrict__ H,
                                                unsigned* __restrict__ slots) {
  __shared__ unsigned rmn[256], rmx[256];
  int b = blockIdx.x, tid = threadIdx.x;
  const float* Hb = H + (size_t)b * NPIX;
  int base = blockIdx.y * 4096;
  unsigned mn = 0xFFFFFFFFu, mx = 0u;
  #pragma unroll 4
  for (int j = 0; j < 16; ++j) {
    unsigned e = f2o(Hb[base + j * 256 + tid]);
    mn = (e < mn) ? e : mn;
    mx = (e > mx) ? e : mx;
  }
  rmn[tid] = mn; rmx[tid] = mx;
  __syncthreads();
  for (int st = 128; st > 0; st >>= 1) {
    if (tid < st) {
      if (rmn[tid + st] < rmn[tid]) rmn[tid] = rmn[tid + st];
      if (rmx[tid + st] > rmx[tid]) rmx[tid] = rmx[tid + st];
    }
    __syncthreads();
  }
  if (tid == 0) {
    atomicMin(&slots[b * 2 + 0], rmn[0]);
    atomicMax(&slots[b * 2 + 1], rmx[0]);
  }
}

// ---------------------------------------------------------------------------
// E = exp(s*H - Xmax), Xmax from stage-0 range. grid (256,16), block 256.
__global__ __launch_bounds__(256) void k_prep(const float* __restrict__ H,
                                              const unsigned* __restrict__ s0,
                                              const float* __restrict__ pr,
                                              float* __restrict__ E) {
  int b = blockIdx.y;
  float s   = pr[b * 52 + 50];
  float omn = o2f(s0[b * 2]), omx = o2f(s0[b * 2 + 1]);
  float Xmax = (s >= 0.0f) ? s * omx : s * omn;
  size_t o = (size_t)b * NPIX + blockIdx.x * 256 + threadIdx.x;
  E[o] = expf(s * H[o] - Xmax);
}

// ---------------------------------------------------------------------------
// morphology layer: Z = s*(log(corr7x7(E, e^w)) + Xmax + bias)
// 16x16 output tile per block, 22x22 halo staged in LDS; padded taps use
// exp(0 - Xmax) (reference zero-pads H before unfold). grid (256,16).
__global__ __launch_bounds__(256) void k_conv(const float* __restrict__ E,
                                              float* __restrict__ Z,
                                              const float* __restrict__ pr,
                                              const unsigned* __restrict__ s0,
                                              unsigned* __restrict__ slotsL) {
  __shared__ float sE[22 * 22];
  __shared__ float sw[KS];
  __shared__ unsigned rmn[256], rmx[256];

  int b   = blockIdx.y;
  int tid = threadIdx.x;
  int x0 = (blockIdx.x & 15) * 16;
  int y0 = (blockIdx.x >> 4) * 16;

  float s    = pr[b * 52 + 50];
  float bias = pr[b * 52 + 49];
  float omn  = o2f(s0[b * 2]), omx = o2f(s0[b * 2 + 1]);
  float Xmax = (s >= 0.0f) ? s * omx : s * omn;
  float epad = expf(-Xmax);

  if (tid < KS) sw[tid] = pr[b * 52 + tid];

  const float* Eb = E + (size_t)b * NPIX;
  for (int i = tid; i < 22 * 22; i += 256) {
    int ly = i / 22, lx = i - ly * 22;
    int gy = y0 + ly - 3, gx = x0 + lx - 3;
    sE[i] = (gx >= 0 && gx < HW && gy >= 0 && gy < HW) ? Eb[gy * HW + gx] : epad;
  }
  __syncthreads();

  int px = tid & 15, py = tid >> 4;
  float c = 0.0f;
  #pragma unroll
  for (int kh = 0; kh < 7; ++kh)
    #pragma unroll
    for (int kw = 0; kw < 7; ++kw)
      c = fmaf(sw[kh * 7 + kw], sE[(py + kh) * 22 + px + kw], c);

  float z = s * (logf(c) + Xmax + bias);
  Z[(size_t)b * NPIX + (y0 + py) * HW + x0 + px] = z;

  unsigned e = f2o(z);
  rmn[tid] = e; rmx[tid] = e;
  __syncthreads();
  for (int st = 128; st > 0; st >>= 1) {
    if (tid < st) {
      if (rmn[tid + st] < rmn[tid]) rmn[tid] = rmn[tid + st];
      if (rmx[tid + st] > rmx[tid]) rmx[tid] = rmx[tid + st];
    }
    __syncthreads();
  }
  if (tid == 0) {
    atomicMin(&slotsL[b * 2 + 0], rmn[0]);
    atomicMax(&slotsL[b * 2 + 1], rmx[0]);
  }
}

// ---------------------------------------------------------------------------
// normalize: H = (Z-vmin)/(vmax-vmin+1e-9)*contrast + ori_min.
// non-final: write next exp-domain image E. final: write H in place (+ scalar).
__global__ __launch_bounds__(256) void k_norm(const float* __restrict__ Z,
                                              const unsigned* __restrict__ s0,
                                              const unsigned* __restrict__ sL,
                                              const float* __restrict__ pr,
                                              float* __restrict__ Eout,
                                              float* __restrict__ Hout,
                                              int is_final) {
  int b = blockIdx.y;
  float omn = o2f(s0[b * 2]), omx = o2f(s0[b * 2 + 1]);
  float vmn = o2f(sL[b * 2]), vmx = o2f(sL[b * 2 + 1]);
  float contrast = omx - omn;
  float inv = 1.0f / (vmx - vmn + 1e-9f);
  size_t o = (size_t)b * NPIX + blockIdx.x * 256 + threadIdx.x;
  float h = (Z[o] - vmn) * inv * contrast + omn;
  if (is_final) {
    Hout[o] = h;
    if (b == 0 && blockIdx.x == 0 && threadIdx.x == 0)
      Hout[(size_t)BATCH * NPIX] = 0.0f;   // loss_reg == 0
  } else {
    float s = pr[b * 52 + 50];
    float Xmax = (s >= 0.0f) ? s * omx : s * omn;
    Eout[o] = expf(s * h - Xmax);
  }
}

// ---------------------------------------------------------------------------
extern "C" void kernel_launch(void* const* d_in, const int* in_sizes, int n_in,
                              void* d_out, int out_size, void* d_ws, size_t ws_size,
                              hipStream_t stream) {
  const float* mask_feats  = (const float*)d_in[0];  // 16x1024x256
  const float* patch_feats = (const float*)d_in[1];  // 16x1024x768
  const float* lrm         = (const float*)d_in[2];  // 16x1x256x256
  const float* W1          = (const float*)d_in[3];  // 1024x512
  const float* b1          = (const float*)d_in[4];  // 512
  const float* W2          = (const float*)d_in[5];  // 512x51
  const float* b2          = (const float*)d_in[6];  // 51
  float* out = (float*)d_out;                        // 16*65536 + 1

  // workspace layout (floats)
  float* ws = (float*)d_ws;
  float*     partial = ws;                             // 8*16*1024 = 131072
  float*     feat    = ws + 131072;                    // 16*1024
  float*     x       = ws + 147456;                    // 16*64
  float*     params  = ws + 148480;                    // 16*52
  unsigned*  slots   = (unsigned*)(ws + 149312);       // 4 stages * 16 * 2
  _Float16*  hbuf    = (_Float16*)(ws + 149504);       // 16*512 f16 (4096 f32)
  float*     bufE    = ws + 163840;                    // 16*65536
  float*     Z       = out;                            // reuse d_out as ping

  k_init<<<1, 128, 0, stream>>>(slots);
  k_meanpool<<<dim3(BATCH, 4, NSPLIT), 256, 0, stream>>>(mask_feats, patch_feats, partial);
  k_reduce_feat<<<64, 256, 0, stream>>>(partial, feat);
  k_mlp1<<<HID / 16, 32, 0, stream>>>(feat, W1, b1, hbuf);
  k_mlp2<<<1, 128, 0, stream>>>(hbuf, W2, b2, x);
  k_params<<<BATCH, 64, 0, stream>>>(x, params);
  k_minmax<<<dim3(BATCH, 16), 256, 0, stream>>>(lrm, slots);           // stage 0
  k_prep<<<dim3(256, BATCH), 256, 0, stream>>>(lrm, slots, params, bufE);

  for (int L = 0; L < 3; ++L) {
    unsigned* sL = slots + (L + 1) * 32;
    k_conv<<<dim3(256, BATCH), 256, 0, stream>>>(bufE, Z, params, slots, sL);
    k_norm<<<dim3(256, BATCH), 256, 0, stream>>>(Z, slots, sL, params,
                                                 bufE, out, (L == 2) ? 1 : 0);
  }
}
